// SimpleAttention_88390426952199
// MI455X (gfx1250) — compile-verified, matrix operand order checked by reading, not run on hardware
//
#include <hip/hip_runtime.h>
#include <hip/hip_bf16.h>

// ---------------------------------------------------------------------------
// Types for CDNA5 WMMA (wave32, 16x16x32 bf16 -> f32)
// ---------------------------------------------------------------------------
typedef __bf16 bf16_t;
typedef __attribute__((ext_vector_type(8)))  __bf16 v8bf;
typedef __attribute__((ext_vector_type(16))) __bf16 v16bf;
typedef __attribute__((ext_vector_type(8)))  float  v8f;

// Problem constants (fixed by the reference)
#define BATCH 4
#define SEQ   2048
#define DMODEL 1024
#define HEADS 16
#define DHEAD 64
// 1/sqrt(64) * log2(e): scores computed in log2 domain -> exp2f softmax
#define QSCALE 0.18033688011112042f

template <bool B> struct BoolC { static constexpr bool value = B; };

#ifndef __has_builtin
#define __has_builtin(x) 0
#endif

// ---------------------------------------------------------------------------
// CDNA5 async global->LDS copies (ASYNCcnt path), guarded so the file always
// compiles; fallback is a plain VGPR-staged copy.
// ---------------------------------------------------------------------------
#if __has_builtin(__builtin_amdgcn_global_load_async_to_lds_b128)
#define HAVE_ASYNC_LDS 1
#else
#define HAVE_ASYNC_LDS 0
#endif

#if HAVE_ASYNC_LDS
typedef int v4i_vs __attribute__((vector_size(16)));

__device__ __forceinline__ void copy16_to_lds(bf16_t* lds_dst, const bf16_t* gsrc) {
  // generic LDS pointer low 32 bits == LDS offset; AS(3) pointers are 32-bit.
  __attribute__((address_space(1))) v4i_vs* gp =
      (__attribute__((address_space(1))) v4i_vs*)(unsigned long long)gsrc;
  __attribute__((address_space(3))) v4i_vs* lp =
      (__attribute__((address_space(3))) v4i_vs*)(unsigned)(unsigned long long)lds_dst;
  __builtin_amdgcn_global_load_async_to_lds_b128(gp, lp, 0, 0);
}
#else
__device__ __forceinline__ void copy16_to_lds(bf16_t* lds_dst, const bf16_t* gsrc) {
  *(v8bf*)lds_dst = *(const v8bf*)gsrc;
}
#endif

__device__ __forceinline__ void wait_async_le8() {
#if HAVE_ASYNC_LDS
  asm volatile("s_wait_asynccnt 0x8" ::: "memory");
#endif
}
__device__ __forceinline__ void wait_async_0() {
#if HAVE_ASYNC_LDS
  asm volatile("s_wait_asynccnt 0x0" ::: "memory");
#endif
}

// ---------------------------------------------------------------------------
// 16-lane row reductions. Prefer v_permlane16_b32 (pure VALU, co-executes
// with WMMA) over __shfl_xor's ds_bpermute (LDS pipe + DScnt waits).
// ---------------------------------------------------------------------------
#if __has_builtin(__builtin_amdgcn_permlane16)
#define HAVE_PERMLANE16 1
__device__ __forceinline__ float permx16(float v, unsigned s0, unsigned s1) {
  int i = __float_as_int(v);
  int r = __builtin_amdgcn_permlane16(i, i, s0, s1, false, false);
  return __int_as_float(r);
}
__device__ __forceinline__ float redmax16(float x) {
  x = fmaxf(x, permx16(x, 0x67452301u, 0xEFCDAB89u));  // xor 1
  x = fmaxf(x, permx16(x, 0x54761032u, 0xDCFE98BAu));  // xor 2
  x = fmaxf(x, permx16(x, 0x32107654u, 0xBA98FEDCu));  // xor 4
  x = fmaxf(x, permx16(x, 0xFEDCBA98u, 0x76543210u));  // xor 8
  return x;
}
__device__ __forceinline__ float redsum16(float x) {
  x += permx16(x, 0x67452301u, 0xEFCDAB89u);
  x += permx16(x, 0x54761032u, 0xDCFE98BAu);
  x += permx16(x, 0x32107654u, 0xBA98FEDCu);
  x += permx16(x, 0xFEDCBA98u, 0x76543210u);
  return x;
}
#else
#define HAVE_PERMLANE16 0
__device__ __forceinline__ float redmax16(float x) {
  x = fmaxf(x, __shfl_xor(x, 1, 32));
  x = fmaxf(x, __shfl_xor(x, 2, 32));
  x = fmaxf(x, __shfl_xor(x, 4, 32));
  x = fmaxf(x, __shfl_xor(x, 8, 32));
  return x;
}
__device__ __forceinline__ float redsum16(float x) {
  x += __shfl_xor(x, 1, 32);
  x += __shfl_xor(x, 2, 32);
  x += __shfl_xor(x, 4, 32);
  x += __shfl_xor(x, 8, 32);
  return x;
}
#endif

__device__ __forceinline__ v16bf make_frag(const bf16_t* p0, const bf16_t* p1) {
  v8bf lo = *(const v8bf*)p0;
  v8bf hi = *(const v8bf*)p1;
  return __builtin_shufflevector(lo, hi, 0,1,2,3,4,5,6,7,8,9,10,11,12,13,14,15);
}

__device__ __forceinline__ v8f wmma_bf16(v16bf a, v16bf b, v8f c) {
  return __builtin_amdgcn_wmma_f32_16x16x32_bf16(
      /*neg_a=*/false, a, /*neg_b=*/false, b,
      /*c_mod=*/(short)0, c, /*reuse_a=*/false, /*reuse_b=*/false);
}

// ---------------------------------------------------------------------------
// Conversion kernels
// ---------------------------------------------------------------------------
__global__ void k_f32_to_bf16(const float* __restrict__ in, bf16_t* __restrict__ out, int n) {
  int idx = blockIdx.x * 256 + threadIdx.x;
  if (idx < n) out[idx] = (bf16_t)in[idx];
}

// in: f32 [K][N] row-major  ->  out: bf16 [N][K] row-major (transposed)
__global__ void k_transpose_to_bf16(const float* __restrict__ in, bf16_t* __restrict__ out,
                                    int K, int N) {
  int idx = blockIdx.x * 256 + threadIdx.x;
  if (idx >= K * N) return;
  int k = idx / N;
  int n = idx - k * N;
  out[(size_t)n * K + k] = (bf16_t)in[idx];
}

// ---------------------------------------------------------------------------
// WMMA GEMM:  C[M,N] = A[M,K] * B^T[N,K]   (bf16 operands, f32 accumulate)
// Block tile 128x128, K staged 64 at a time, double-buffered LDS filled by
// async global->LDS copies (8 per thread per stage -> s_wait_asynccnt 8).
// 8 waves: 2(M) x 4(N); each wave owns 4x2 tiles of 16x16 -> 16 WMMA/stage.
// EPI=0: scatter epilogue into q (scaled), k, v^T buffers.  EPI=1: f32 store.
// ---------------------------------------------------------------------------
#define BM 128
#define BN 128
#define BK 64
#define LDT (BK + 8)   // padded LDS row stride: 144 B, 16B-aligned

template <int EPI>
__global__ __launch_bounds__(256)
void k_gemm_bf16(const bf16_t* __restrict__ A, const bf16_t* __restrict__ BT,
                 int M, int N, int K,
                 float* __restrict__ fout,
                 bf16_t* __restrict__ qbuf, bf16_t* __restrict__ kbuf,
                 bf16_t* __restrict__ vtbuf) {
  __shared__ __attribute__((aligned(16))) bf16_t As[2][BM * LDT];
  __shared__ __attribute__((aligned(16))) bf16_t Bs[2][BN * LDT];

  const int tid  = threadIdx.x;
  const int lane = tid & 31;
  const int wid  = tid >> 5;
  const int wm   = wid & 1;       // 0..1 (M): 64 rows each
  const int wn   = wid >> 1;      // 0..3 (N): 32 cols each
  const int m0   = blockIdx.x * BM;
  const int n0   = blockIdx.y * BN;

  const int mf  = lane & 15;            // fragment row/col within 16-tile
  const int kb  = (lane >> 4) << 3;     // 0 or 8 : K sub-offset of fragment
  const int rhi = (lane >> 4) << 3;     // 0 or 8 : C/D row offset

  v8f acc[4][2] = {};

  auto stage = [&](int buf, int k0) {
#pragma unroll
    for (int i = 0; i < 4; ++i) {       // A tile: 128x64 bf16 = 1024 chunks
      int idx = tid + i * 256;
      int row = idx >> 3;
      int c8  = (idx & 7) << 3;
      copy16_to_lds(&As[buf][row * LDT + c8],
                    &A[(size_t)(m0 + row) * K + k0 + c8]);
    }
#pragma unroll
    for (int i = 0; i < 4; ++i) {       // B tile (N-major): 128x64 bf16
      int idx = tid + i * 256;
      int row = idx >> 3;
      int c8  = (idx & 7) << 3;
      copy16_to_lds(&Bs[buf][row * LDT + c8],
                    &BT[(size_t)(n0 + row) * K + k0 + c8]);
    }
  };

  const int nstage = K / BK;
  stage(0, 0);

  for (int s = 0; s < nstage; ++s) {
    const int cur  = s & 1;
    const bool more = (s + 1) < nstage;
    if (more) {
      stage(cur ^ 1, (s + 1) * BK);
      wait_async_le8();                 // previous stage's 8 copies complete
    } else {
      wait_async_0();
    }
    __syncthreads();

#pragma unroll
    for (int kc = 0; kc < BK; kc += 32) {
      v16bf af[4], bf[2];
#pragma unroll
      for (int i = 0; i < 4; ++i) {
        const bf16_t* pa = &As[cur][(wm * 64 + i * 16 + mf) * LDT + kc + kb];
        af[i] = make_frag(pa, pa + 16);
      }
#pragma unroll
      for (int j = 0; j < 2; ++j) {
        const bf16_t* pb = &Bs[cur][(wn * 32 + j * 16 + mf) * LDT + kc + kb];
        bf[j] = make_frag(pb, pb + 16);
      }
#pragma unroll
      for (int i = 0; i < 4; ++i)
#pragma unroll
        for (int j = 0; j < 2; ++j)
          acc[i][j] = wmma_bf16(af[i], bf[j], acc[i][j]);
    }
    __syncthreads();
  }

  // ---- epilogue ----
#pragma unroll
  for (int i = 0; i < 4; ++i) {
#pragma unroll
    for (int j = 0; j < 2; ++j) {
#pragma unroll
      for (int r = 0; r < 8; ++r) {
        int gm = m0 + wm * 64 + i * 16 + rhi + r;
        int gn = n0 + wn * 32 + j * 16 + mf;
        float v = acc[i][j][r];
        if (EPI == 1) {
          fout[(size_t)gm * N + gn] = v;
        } else {
          int section = gn >> 10;       // 0=Q, 1=K, 2=V
          int dcol    = gn & 1023;
          int h  = dcol >> 6;
          int di = dcol & 63;
          int b  = gm >> 11;
          int l  = gm & 2047;
          size_t bh = (size_t)(b * HEADS + h);
          if (section == 0) {
            qbuf[(bh * SEQ + l) * DHEAD + di] = (bf16_t)(v * QSCALE);
          } else if (section == 1) {
            kbuf[(bh * SEQ + l) * DHEAD + di] = (bf16_t)v;
          } else {
            vtbuf[(bh * DHEAD + di) * SEQ + l] = (bf16_t)v;           // V transposed
          }
        }
      }
    }
  }
}

// ---------------------------------------------------------------------------
// Flash attention (causal). One wave per 16 query rows; 8 waves / block.
// Scores are in log2 domain (QSCALE includes log2 e) -> exp2f softmax.
// Key loop split: mask-free fast path for tiles fully below the diagonal,
// compare/cndmask masking only on the 1-2 tiles straddling it.
// Row reductions via v_permlane16_b32 (VALU) instead of ds_bpermute.
// ---------------------------------------------------------------------------
#define PSTRIDE 48   // 96 B row stride, 16B aligned for ds_read_b128

__global__ __launch_bounds__(256)
void k_flash_attn(const bf16_t* __restrict__ q, const bf16_t* __restrict__ k,
                  const bf16_t* __restrict__ vt, bf16_t* __restrict__ aout) {
  __shared__ __attribute__((aligned(16))) bf16_t pbuf[8][16 * PSTRIDE];

  const int tid  = threadIdx.x;
  const int lane = tid & 31;
  const int wid  = tid >> 5;

  const int bx = blockIdx.x;          // = ((b*16 + h)*16 + qblock)
  const int qb = bx & 15;
  const int h  = (bx >> 4) & 15;
  const int b  = bx >> 8;

  const int q0 = qb * 128 + wid * 16;
  const size_t bh = (size_t)(b * HEADS + h);
  const bf16_t* qbh = q  + bh * SEQ * DHEAD;
  const bf16_t* kbh = k  + bh * SEQ * DHEAD;
  const bf16_t* vbh = vt + bh * DHEAD * SEQ;

  const int mf  = lane & 15;
  const int kb  = (lane >> 4) << 3;
  const int rhi = (lane >> 4) << 3;

  // Q fragments (A-matrix layout), dh=64 -> two K=32 chunks; q pre-scaled.
  v16bf qf[2];
#pragma unroll
  for (int c = 0; c < 2; ++c) {
    const bf16_t* p = &qbh[(size_t)(q0 + mf) * DHEAD + c * 32 + kb];
    qf[c] = make_frag(p, p + 16);
  }

  v8f o[4] = {};
  float mrow[8], lrow[8];
#pragma unroll
  for (int r = 0; r < 8; ++r) { mrow[r] = -3.0e38f; lrow[r] = 0.0f; }

  bf16_t* pw = &pbuf[wid][0];
  const int kmax = q0 + 15;

  auto tile = [&](int kt, auto masked_c) {
    constexpr bool MASKED = decltype(masked_c)::value;

    // ---- prefetch next 32-key tile of K and V ----
    if (kt + 32 <= kmax) {
      __builtin_prefetch(&kbh[(size_t)(kt + 32 + lane) * DHEAD], 0, 3);
      __builtin_prefetch(&vbh[(size_t)lane * SEQ + kt + 32], 0, 3);
      __builtin_prefetch(&vbh[(size_t)(32 + lane) * SEQ + kt + 32], 0, 3);
    }

    // ---- scores: two 16x16 D tiles covering keys kt..kt+31 ----
    v8f s[2] = {};
#pragma unroll
    for (int t = 0; t < 2; ++t) {
#pragma unroll
      for (int c = 0; c < 2; ++c) {
        const bf16_t* pk = &kbh[(size_t)(kt + t * 16 + mf) * DHEAD + c * 32 + kb];
        v16bf kf = make_frag(pk, pk + 16);
        s[t] = wmma_bf16(qf[c], kf, s[t]);
      }
    }

    if (MASKED) {
#pragma unroll
      for (int t = 0; t < 2; ++t) {
#pragma unroll
        for (int r = 0; r < 8; ++r) {
          int gm = q0 + rhi + r;
          int gn = kt + t * 16 + mf;
          if (gn > gm) s[t][r] = -3.0e38f;
        }
      }
    }

    // ---- online softmax (exp2 domain); stash P (bf16) into LDS ----
#pragma unroll
    for (int r = 0; r < 8; ++r) {
      float tmax = redmax16(fmaxf(s[0][r], s[1][r]));
      float mnew  = fmaxf(mrow[r], tmax);
      float alpha = exp2f(mrow[r] - mnew);
      float p0 = exp2f(s[0][r] - mnew);
      float p1 = exp2f(s[1][r] - mnew);
      float rs = redsum16(p0 + p1);
      lrow[r] = lrow[r] * alpha + rs;
      mrow[r] = mnew;
#pragma unroll
      for (int j = 0; j < 4; ++j) o[j][r] *= alpha;
      pw[(rhi + r) * PSTRIDE + mf]      = (bf16_t)p0;
      pw[(rhi + r) * PSTRIDE + 16 + mf] = (bf16_t)p1;
    }

    // ---- re-load P in A-fragment layout (same-wave DS ops in-order) ----
    v16bf pf = make_frag(&pw[mf * PSTRIDE + kb], &pw[mf * PSTRIDE + kb + 16]);

    // ---- O += P * V  (V stored transposed: [dh][L]) ----
#pragma unroll
    for (int j = 0; j < 4; ++j) {
      const bf16_t* pv = &vbh[(size_t)(j * 16 + mf) * SEQ + kt + kb];
      v16bf vf = make_frag(pv, pv + 16);
      o[j] = wmma_bf16(pf, vf, o[j]);
    }
  };

  // tiles fully below the diagonal (kt+31 <= q0): no masking needed
  const int kt_split = (q0 >= 31) ? ((((q0 - 31) >> 5) << 5) + 32) : 0;
  for (int kt = 0; kt < kt_split; kt += 32) tile(kt, BoolC<false>{});
  for (int kt = kt_split; kt <= kmax; kt += 32) tile(kt, BoolC<true>{});

  // ---- normalize + write attention output (bf16, row-major [B*L, D]) ----
#pragma unroll
  for (int r = 0; r < 8; ++r) {
    float linv = 1.0f / lrow[r];
    size_t row = (size_t)b * SEQ + q0 + rhi + r;
#pragma unroll
    for (int j = 0; j < 4; ++j) {
      int col = h * DHEAD + j * 16 + mf;
      aout[row * DMODEL + col] = (bf16_t)(o[j][r] * linv);
    }
  }
}

// ---------------------------------------------------------------------------
// Host launcher
// ---------------------------------------------------------------------------
extern "C" void kernel_launch(void* const* d_in, const int* in_sizes, int n_in,
                              void* d_out, int out_size, void* d_ws, size_t ws_size,
                              hipStream_t stream) {
  (void)in_sizes; (void)n_in; (void)out_size; (void)ws_size;
  const float* x     = (const float*)d_in[0];   // [4,2048,1024]
  const float* w_qkv = (const float*)d_in[1];   // [1024,3072]
  const float* w_out = (const float*)d_in[2];   // [1024,1024]
  float* out = (float*)d_out;                   // [4,2048,1024]

  const int M  = BATCH * SEQ;                   // 8192
  const int K  = DMODEL;                        // 1024
  const int N3 = 3 * DMODEL;                    // 3072

  char* ws = (char*)d_ws;
  size_t off = 0;
  bf16_t* x_bf   = (bf16_t*)(ws + off); off += (size_t)M * K * 2;        // 16 MB
  bf16_t* wqkvT  = (bf16_t*)(ws + off); off += (size_t)N3 * K * 2;       //  6 MB
  bf16_t* woutT  = (bf16_t*)(ws + off); off += (size_t)DMODEL * K * 2;   //  2 MB
  bf16_t* qb     = (bf16_t*)(ws + off); off += (size_t)M * DMODEL * 2;   // 16 MB
  bf16_t* kbuf   = (bf16_t*)(ws + off); off += (size_t)M * DMODEL * 2;   // 16 MB
  bf16_t* vtb    = (bf16_t*)(ws + off); off += (size_t)M * DMODEL * 2;   // 16 MB
  bf16_t* aoutb  = (bf16_t*)(ws + off); off += (size_t)M * DMODEL * 2;   // 16 MB

  // 1) precision conversion / weight transposition
  k_f32_to_bf16<<<(M * K) / 256, 256, 0, stream>>>(x, x_bf, M * K);
  k_transpose_to_bf16<<<(K * N3) / 256, 256, 0, stream>>>(w_qkv, wqkvT, K, N3);
  k_transpose_to_bf16<<<(K * DMODEL) / 256, 256, 0, stream>>>(w_out, woutT, K, DMODEL);

  // 2) fused QKV projection (WMMA GEMM), scatters q(scaled)/k/v^T
  dim3 g1(M / BM, N3 / BN);
  k_gemm_bf16<0><<<g1, 256, 0, stream>>>(x_bf, wqkvT, M, N3, K,
                                         nullptr, qb, kbuf, vtb);

  // 3) causal flash attention
  k_flash_attn<<<BATCH * HEADS * (SEQ / 128), 256, 0, stream>>>(qb, kbuf, vtb, aoutb);

  // 4) output projection (WMMA GEMM, f32 epilogue)
  dim3 g2(M / BM, DMODEL / BN);
  k_gemm_bf16<1><<<g2, 256, 0, stream>>>(aoutb, woutT, M, DMODEL, K,
                                         out, nullptr, nullptr, nullptr);
}
